// MultiHeadAttention_36069135351847
// MI455X (gfx1250) — matrix operand order repaired and checked by reading, not compiled, not run on hardware
//
#include <hip/hip_runtime.h>
#include <stdint.h>

// ---------------- problem constants ----------------
#define BB 2
#define SS 2048
#define DD 1024
#define HH 16
#define DHH 64

// ---------------- vector types ----------------
typedef __attribute__((ext_vector_type(16))) __bf16          bf16x16;
typedef __attribute__((ext_vector_type(8)))  float           f32x8;
typedef __attribute__((ext_vector_type(16))) unsigned short  u16x16;
typedef __attribute__((ext_vector_type(8)))  unsigned short  u16x8;
typedef __attribute__((ext_vector_type(4)))  int             i32x4;

#if __has_builtin(__builtin_amdgcn_global_load_async_to_lds_b128) && \
    __has_builtin(__builtin_amdgcn_s_wait_asynccnt)
#define HAVE_ASYNC_LDS 1
#else
#define HAVE_ASYNC_LDS 0
#endif

__device__ __forceinline__ unsigned short f32_bf16(float f) {
    union { float f; uint32_t u; } c; c.f = f;
    uint32_t u = c.u;
    uint32_t r = u + 0x7FFFu + ((u >> 16) & 1u);   // round-to-nearest-even
    return (unsigned short)(r >> 16);
}

// A fragment (16x32 bf16, row-major source, lda in elements).
__device__ __forceinline__ bf16x16 load_a_bf16(const unsigned short* A, int lda,
                                               int row0, int k0, int lane) {
    int r  = lane & 15;
    int kb = k0 + ((lane >> 4) << 4);
    const unsigned short* p = A + (size_t)(row0 + r) * lda + kb;
    u16x8 lo = *reinterpret_cast<const u16x8*>(p);
    u16x8 hi = *reinterpret_cast<const u16x8*>(p + 8);
    u16x16 v;
#pragma unroll
    for (int i = 0; i < 8; ++i) { v[i] = lo[i]; v[i + 8] = hi[i]; }
    return __builtin_bit_cast(bf16x16, v);
}

// B fragment (32x16 bf16) from an [N,K] row-major source (ldb in elements).
__device__ __forceinline__ bf16x16 load_b_bf16(const unsigned short* Bm, int ldb,
                                               int col0, int k0, int lane) {
    int c  = lane & 15;
    int kb = k0 + ((lane >> 4) << 4);
    const unsigned short* p = Bm + (size_t)(col0 + c) * ldb + kb;
    u16x16 v = *reinterpret_cast<const u16x16*>(p);
    return __builtin_bit_cast(bf16x16, v);
}

__device__ __forceinline__ f32x8 wmma_bf16(bf16x16 a, bf16x16 b, f32x8 c) {
    return __builtin_amdgcn_wmma_f32_16x16x32_bf16(
        /*neg_a=*/false, a, /*neg_b=*/false, b,
        /*c_mod=*/(short)0, c, /*reuse_a=*/false, /*reuse_b=*/false);
}

// async global -> LDS copy of 16 bytes (gfx1250 GLOBAL_LOAD_ASYNC_TO_LDS_B128)
__device__ __forceinline__ void async_g2l_b128(const void* g, void* l) {
#if HAVE_ASYNC_LDS
    __builtin_amdgcn_global_load_async_to_lds_b128(
        (__attribute__((address_space(1))) i32x4*)g,
        (__attribute__((address_space(3))) i32x4*)l, 0, 0);
#else
    *(int4*)l = *(const int4*)g;
#endif
}

__device__ __forceinline__ void async_wait0() {
#if HAVE_ASYNC_LDS
    __builtin_amdgcn_s_wait_asynccnt(0);
#endif
}

// ---------------- f32 -> bf16 convert ----------------
__global__ void cvt_kernel(const float* __restrict__ src,
                           unsigned short* __restrict__ dst, int n) {
    int i = blockIdx.x * blockDim.x + threadIdx.x;
    if (i < n) dst[i] = f32_bf16(src[i]);
}

// ---------------- WMMA GEMM: C = A[M,K] * Bw[N,K]^T + bias ----------------
// Wave tile 32x64 (2 A-frags x 4 B-frags, 8 WMMA per k-step), block 64x256.
// mode 0: bf16 row-major out; mode 1: bf16 transposed into Vt[B,H,DH,S];
// mode 2: f32 row-major out.
__global__ __launch_bounds__(256) void gemm_kernel(
    const unsigned short* __restrict__ A, const unsigned short* __restrict__ Bw,
    const float* __restrict__ bias, unsigned short* __restrict__ outB,
    float* __restrict__ outF, int M, int N, int K, int mode) {
    int lane = threadIdx.x & 31;
    int wave = threadIdx.x >> 5;
    int wm = wave >> 2;                 // 0..1
    int wn = wave & 3;                  // 0..3
    int m0 = blockIdx.y * 64  + wm * 32;
    int n0 = blockIdx.x * 256 + wn * 64;

    f32x8 acc[2][4] = {};
    for (int k0 = 0; k0 < K; k0 += 32) {
        bf16x16 a0 = load_a_bf16(A, K, m0,      k0, lane);
        bf16x16 a1 = load_a_bf16(A, K, m0 + 16, k0, lane);
        bf16x16 bfr[4];
#pragma unroll
        for (int j = 0; j < 4; ++j)
            bfr[j] = load_b_bf16(Bw, K, n0 + j * 16, k0, lane);
#pragma unroll
        for (int j = 0; j < 4; ++j) {
            acc[0][j] = wmma_bf16(a0, bfr[j], acc[0][j]);
            acc[1][j] = wmma_bf16(a1, bfr[j], acc[1][j]);
        }
    }

    int col = lane & 15, half = lane >> 4;
#pragma unroll
    for (int tm = 0; tm < 2; ++tm) {
#pragma unroll
        for (int tn = 0; tn < 4; ++tn) {
            int n = n0 + tn * 16 + col;
            float bv = bias[n];
#pragma unroll
            for (int r = 0; r < 8; ++r) {
                int m = m0 + tm * 16 + half * 8 + r;
                float v = acc[tm][tn][r] + bv;
                if (mode == 0) {
                    outB[(size_t)m * N + n] = f32_bf16(v);
                } else if (mode == 1) {
                    int bi = m >> 11, s = m & (SS - 1);
                    int h = n >> 6, dh = n & (DHH - 1);
                    outB[((((size_t)bi * HH + h) * DHH + dh) << 11) + s] = f32_bf16(v);
                } else {
                    outF[(size_t)m * N + n] = v;
                }
            }
        }
    }
}

// ---------------- flash attention: 1 wave per 16-row q-tile per (b,h) ----
__global__ __launch_bounds__(32) void flash_kernel(
    const unsigned short* __restrict__ Q, const unsigned short* __restrict__ Km,
    const unsigned short* __restrict__ Vt, const int* __restrict__ mask,
    unsigned short* __restrict__ Ctx) {
    __shared__ unsigned short pl[16 * 32];   // P tile (bf16)
    __shared__ int            mlds[16 * 32]; // mask tile

    int lane = threadIdx.x;
    int q0 = blockIdx.x * 16;
    int h  = blockIdx.y;
    int b  = blockIdx.z;
    int col = lane & 15, half = lane >> 4;

    // resident Q fragments: 16 x 64 split into two 16x32 A-frags
    bf16x16 aq0 = load_a_bf16(Q, DD, b * SS + q0, h * DHH + 0,  lane);
    bf16x16 aq1 = load_a_bf16(Q, DD, b * SS + q0, h * DHH + 32, lane);

    f32x8 o[4] = {};
    float mrow[8], lrow[8];
#pragma unroll
    for (int r = 0; r < 8; ++r) { mrow[r] = -1e30f; lrow[r] = 0.0f; }

    for (int kv0 = 0; kv0 < SS; kv0 += 32) {
        // stage 16x32 mask tile into LDS via async copies (4 x b128 per lane)
#pragma unroll
        for (int t = 0; t < 4; ++t) {
            int c = t * 32 + lane;          // 0..127 chunk id
            int row = c >> 3, seg = c & 7;  // row 0..15, 16B segment 0..7
            const int* g = mask + ((size_t)b * SS + q0 + row) * SS + kv0 + seg * 4;
            async_g2l_b128(g, &mlds[row * 32 + seg * 4]);
        }

        // scores: two 16x16 tiles (cols kv0..+15, kv0+16..+31), K=64
        f32x8 s0 = {}, s1 = {};
        bf16x16 bk;
        bk = load_b_bf16(Km, DD, b * SS + kv0,      h * DHH + 0,  lane);
        s0 = wmma_bf16(aq0, bk, s0);
        bk = load_b_bf16(Km, DD, b * SS + kv0,      h * DHH + 32, lane);
        s0 = wmma_bf16(aq1, bk, s0);
        bk = load_b_bf16(Km, DD, b * SS + kv0 + 16, h * DHH + 0,  lane);
        s1 = wmma_bf16(aq0, bk, s1);
        bk = load_b_bf16(Km, DD, b * SS + kv0 + 16, h * DHH + 32, lane);
        s1 = wmma_bf16(aq1, bk, s1);

        async_wait0();
        __syncthreads();

        // scale + mask + online softmax (rows live in 16-lane halves)
#pragma unroll
        for (int r = 0; r < 8; ++r) {
            int m = half * 8 + r;
            float x0 = s0[r] * 0.125f;           // 1/sqrt(64)
            float x1 = s1[r] * 0.125f;
            if (mlds[m * 32 + col]      == 0) x0 = -50000.0f;
            if (mlds[m * 32 + 16 + col] == 0) x1 = -50000.0f;
            float mx = fmaxf(x0, x1);
#pragma unroll
            for (int off = 8; off >= 1; off >>= 1)
                mx = fmaxf(mx, __shfl_xor(mx, off, 32));
            float mnew = fmaxf(mrow[r], mx);
            float p0 = __expf(x0 - mnew);
            float p1 = __expf(x1 - mnew);
            float corr = __expf(mrow[r] - mnew);
            float rs = p0 + p1;
#pragma unroll
            for (int off = 8; off >= 1; off >>= 1)
                rs += __shfl_xor(rs, off, 32);
            lrow[r] = lrow[r] * corr + rs;
            mrow[r] = mnew;
#pragma unroll
            for (int g = 0; g < 4; ++g) o[g][r] *= corr;
            pl[m * 32 + col]      = f32_bf16(p0);
            pl[m * 32 + 16 + col] = f32_bf16(p1);
        }
        __syncthreads();

        // P (16x32) as A-fragment from LDS; V via Vt (contiguous per lane)
        {
            int rr = lane & 15;
            int kb = (lane >> 4) << 4;
            const unsigned short* pp = &pl[rr * 32 + kb];
            u16x8 lo = *reinterpret_cast<const u16x8*>(pp);
            u16x8 hi = *reinterpret_cast<const u16x8*>(pp + 8);
            u16x16 v;
#pragma unroll
            for (int i = 0; i < 8; ++i) { v[i] = lo[i]; v[i + 8] = hi[i]; }
            bf16x16 ap = __builtin_bit_cast(bf16x16, v);
#pragma unroll
            for (int g = 0; g < 4; ++g) {
                bf16x16 bv = load_b_bf16(Vt, SS, (b * HH + h) * DHH + g * 16,
                                         kv0, lane);
                o[g] = wmma_bf16(ap, bv, o[g]);
            }
        }
        __syncthreads();
    }

    // normalize and store context (bf16, [B*S, D] with head offset)
#pragma unroll
    for (int g = 0; g < 4; ++g) {
        int dh = g * 16 + col;
#pragma unroll
        for (int r = 0; r < 8; ++r) {
            int q = q0 + half * 8 + r;
            float v = o[g][r] / lrow[r];
            Ctx[((size_t)(b * SS + q)) * DD + h * DHH + dh] = f32_bf16(v);
        }
    }
}

// ---------------- host side ----------------
extern "C" void kernel_launch(void* const* d_in, const int* in_sizes, int n_in,
                              void* d_out, int out_size, void* d_ws, size_t ws_size,
                              hipStream_t stream) {
    const float* x  = (const float*)d_in[0];
    const float* y  = (const float*)d_in[1];
    const int*  msk = (const int*)  d_in[2];
    const float* Wq = (const float*)d_in[3];
    const float* bq = (const float*)d_in[4];
    const float* Wk = (const float*)d_in[5];
    const float* bk = (const float*)d_in[6];
    const float* Wv = (const float*)d_in[7];
    const float* bv = (const float*)d_in[8];
    const float* Wo = (const float*)d_in[9];
    const float* bo = (const float*)d_in[10];
    float* out = (float*)d_out;

    const size_t MD = (size_t)BB * SS * DD;  // 4M elements
    const size_t WW = (size_t)DD * DD;       // 1M elements

    char* ws = (char*)d_ws;
    unsigned short* Xb  = (unsigned short*)ws; ws += MD * 2;
    unsigned short* Yb  = (unsigned short*)ws; ws += MD * 2;
    unsigned short* Wqb = (unsigned short*)ws; ws += WW * 2;
    unsigned short* Wkb = (unsigned short*)ws; ws += WW * 2;
    unsigned short* Wvb = (unsigned short*)ws; ws += WW * 2;
    unsigned short* Wob = (unsigned short*)ws; ws += WW * 2;
    unsigned short* Qb  = (unsigned short*)ws; ws += MD * 2;
    unsigned short* Kb  = (unsigned short*)ws; ws += MD * 2;
    unsigned short* Vt  = (unsigned short*)ws; ws += MD * 2;
    unsigned short* Ctx = (unsigned short*)ws; ws += MD * 2;

    // bf16 conversions
    cvt_kernel<<<(int)((MD + 255) / 256), 256, 0, stream>>>(x,  Xb,  (int)MD);
    cvt_kernel<<<(int)((MD + 255) / 256), 256, 0, stream>>>(y,  Yb,  (int)MD);
    cvt_kernel<<<(int)((WW + 255) / 256), 256, 0, stream>>>(Wq, Wqb, (int)WW);
    cvt_kernel<<<(int)((WW + 255) / 256), 256, 0, stream>>>(Wk, Wkb, (int)WW);
    cvt_kernel<<<(int)((WW + 255) / 256), 256, 0, stream>>>(Wv, Wvb, (int)WW);
    cvt_kernel<<<(int)((WW + 255) / 256), 256, 0, stream>>>(Wo, Wob, (int)WW);

    // projections (block tile 64x256)
    dim3 gg(DD / 256, (BB * SS) / 64);
    gemm_kernel<<<gg, 256, 0, stream>>>(Xb, Wqb, bq, Qb, nullptr,
                                        BB * SS, DD, DD, 0);
    gemm_kernel<<<gg, 256, 0, stream>>>(Yb, Wkb, bk, Kb, nullptr,
                                        BB * SS, DD, DD, 0);
    gemm_kernel<<<gg, 256, 0, stream>>>(Yb, Wvb, bv, Vt, nullptr,
                                        BB * SS, DD, DD, 1);  // transposed V

    // flash attention
    dim3 gf(SS / 16, HH, BB);
    flash_kernel<<<gf, 32, 0, stream>>>(Qb, Kb, Vt, msk, Ctx);

    // output projection -> f32 d_out
    gemm_kernel<<<gg, 256, 0, stream>>>(Ctx, Wob, bo, nullptr, out,
                                        BB * SS, DD, DD, 2);
    (void)in_sizes; (void)n_in; (void)out_size; (void)ws_size;
}
